// mamba_reshape_53317724013020
// MI455X (gfx1250) — compile-verified, matrix operand order checked by reading, not run on hardware
//
#include <hip/hip_runtime.h>
#include <hip/hip_bf16.h>

typedef __attribute__((ext_vector_type(16))) _Float16 v16h;
typedef __attribute__((ext_vector_type(8)))  _Float16 v8h;
typedef __attribute__((ext_vector_type(8)))  float    v8f;

#define T_TOKENS 131072   // 2048 sequences (bb) * 64 steps (s); token = s*2048 + bb
#define D_STATE  16

__device__ __forceinline__ float silu_f(float x) {
    return x / (1.f + __expf(-x));
}
__device__ __forceinline__ float softplus_f(float x) {
    return (x > 20.f) ? x : __logf(1.f + __expf(x));
}
// K index for element e of a 16-bit 16x32 A/B fragment (kb = 0 or 8 per lane half)
__device__ __forceinline__ int frag_k(int kb, int e) {
    return kb + ((e < 8) ? e : (e + 8));
}

// ---- K1: in_proj [T,32] x [32,128] -> xh[T,64], z[T,64] (f16, [s][bb] order) --
// One wave handles 4 M-tiles (64 consecutive tokens) reusing 8 weight fragments.
__global__ void mamba_k1_inproj(const float* __restrict__ x,
                                const float* __restrict__ w,   // (128,32) row-major
                                _Float16* __restrict__ xh,
                                _Float16* __restrict__ zb) {
    __shared__ __align__(16) _Float16 stg[8][16 * 128];
    const int wv    = threadIdx.x >> 5;
    const int lane  = threadIdx.x & 31;
    const int m16   = lane & 15;
    const int hi    = lane >> 4;
    const int kb    = hi ? 8 : 0;
    const int gwave = blockIdx.x * 8 + wv;      // 0..2047
    const long tok0 = (long)gwave * 64;         // 64 tokens, never straddles s

    // Weight B-fragments, loaded once, reused for 4 M-tiles (32 WMMAs)
    v16h bf[8];
    #pragma unroll
    for (int j = 0; j < 8; ++j) {
        #pragma unroll
        for (int e = 0; e < 16; ++e) {
            int k  = frag_k(kb, e);
            int ng = j * 16 + m16;              // B fragment: n = lane&15
            bf[j][e] = (_Float16)w[ng * 32 + k];
        }
    }

    for (int mt = 0; mt < 4; ++mt) {
        const long t0  = tok0 + mt * 16;
        const long tok = t0 + m16;              // this lane's A row
        const int  s   = (int)(tok >> 11);
        const int  bbi = (int)(tok & 2047);
        const int  b   = bbi >> 10;
        const int  hw  = bbi & 1023;
        // x[b][s][k][h][w]; consecutive rows (m) -> consecutive hw -> coalesced
        const long base = ((long)(b * 64 + s) * 32) * 1024 + hw;

        v16h a;
        #pragma unroll
        for (int e = 0; e < 16; ++e) {
            int k = frag_k(kb, e);
            a[e] = (_Float16)x[base + (long)k * 1024];
        }
        #pragma unroll
        for (int j = 0; j < 8; ++j) {
            v8f c = {};
            c = __builtin_amdgcn_wmma_f32_16x16x32_f16(false, a, false, bf[j],
                                                       (short)0, c, false, false);
            #pragma unroll
            for (int v = 0; v < 8; ++v) {
                int mm  = v + hi * 8;
                int col = j * 16 + m16;
                stg[wv][mm * 128 + col] = (_Float16)c[v];
            }
        }
        __syncthreads();
        // coalesced b128 stores: 16 rows x 16 segments of 8 f16
        for (int i = lane; i < 256; i += 32) {
            int row  = i >> 4;
            int seg  = i & 15;
            float4 vd = *(const float4*)&stg[wv][row * 128 + seg * 8];
            long trow = t0 + row;
            int  col0 = seg * 8;
            if (col0 < 64) *(float4*)&xh[trow * 64 + col0] = vd;
            else           *(float4*)&zb[trow * 64 + (col0 - 64)] = vd;
        }
        __syncthreads();
    }
}

// ---- K2: causal depthwise conv(4) + SiLU -> xt (f16) -------------------------
__global__ void mamba_k2_conv(const _Float16* __restrict__ xh,
                              const float* __restrict__ cw,   // (64,1,4)
                              const float* __restrict__ cb,   // (64,)
                              _Float16* __restrict__ xt) {
    const int idx = blockIdx.x * blockDim.x + threadIdx.x;    // bb*64 + d
    const int bb = idx >> 6, d = idx & 63;
    const float w0 = cw[d * 4 + 0], w1 = cw[d * 4 + 1];
    const float w2 = cw[d * 4 + 2], w3 = cw[d * 4 + 3];
    const float bias = cb[d];
    float x0 = 0.f, x1 = 0.f, x2 = 0.f;
    for (int t = 0; t < 64; ++t) {
        const long tok = (long)t * 2048 + bb;
        float x3 = (float)xh[tok * 64 + d];
        float acc = bias + w0 * x0 + w1 * x1 + w2 * x2 + w3 * x3;
        xt[tok * 64 + d] = (_Float16)silu_f(acc);
        x0 = x1; x1 = x2; x2 = x3;
    }
}

// ---- K3: x_proj [T,64]x[64,34] -> dt(f32), B(f16), C(f16) --------------------
__global__ void mamba_k3_xproj(const _Float16* __restrict__ xt,
                               const float* __restrict__ xw,   // (34,64)
                               _Float16* __restrict__ Bm,      // [T,16]
                               _Float16* __restrict__ Cm,      // [T,16]
                               float* __restrict__ dt2) {      // [T,2]
    __shared__ __align__(16) _Float16 stg[8][16 * 32];         // per row: B(16)|C(16)
    const int wv    = threadIdx.x >> 5;
    const int lane  = threadIdx.x & 31;
    const int m16   = lane & 15;
    const int hi    = lane >> 4;
    const int kb    = hi ? 8 : 0;
    const int gwave = blockIdx.x * 8 + wv;                     // tile 0..8191
    const long token = (long)gwave * 16 + m16;

    v16h a0, a1;
    #pragma unroll
    for (int e = 0; e < 16; ++e) {
        int k = frag_k(kb, e);
        a0[e] = xt[token * 64 + k];
        a1[e] = xt[token * 64 + 32 + k];
    }
    #pragma unroll
    for (int j = 0; j < 3; ++j) {                              // N padded 34 -> 48
        v16h b0, b1;
        #pragma unroll
        for (int e = 0; e < 16; ++e) {
            int k  = frag_k(kb, e);
            int ng = j * 16 + m16;
            b0[e] = (ng < 34) ? (_Float16)xw[ng * 64 + k]      : (_Float16)0.f;
            b1[e] = (ng < 34) ? (_Float16)xw[ng * 64 + 32 + k] : (_Float16)0.f;
        }
        v8f c = {};
        c = __builtin_amdgcn_wmma_f32_16x16x32_f16(false, a0, false, b0,
                                                   (short)0, c, false, false);
        c = __builtin_amdgcn_wmma_f32_16x16x32_f16(false, a1, false, b1,
                                                   (short)0, c, false, false);
        #pragma unroll
        for (int v = 0; v < 8; ++v) {
            int mm  = v + hi * 8;
            int col = j * 16 + m16;
            if (col < 2)       dt2[((long)gwave * 16 + mm) * 2 + col] = c[v];
            else if (col < 18) stg[wv][mm * 32 + (col - 2)]       = (_Float16)c[v];
            else if (col < 34) stg[wv][mm * 32 + 16 + (col - 18)] = (_Float16)c[v];
        }
    }
    __syncthreads();
    for (int i = lane; i < 64; i += 32) {                      // 16 rows x 4 segs
        int row = i >> 2, seg = i & 3;
        float4 vd = *(const float4*)&stg[wv][row * 32 + seg * 8];
        long tok = (long)gwave * 16 + row;
        if (seg < 2) *(float4*)&Bm[tok * 16 + seg * 8]       = vd;
        else         *(float4*)&Cm[tok * 16 + (seg - 2) * 8] = vd;
    }
}

// ---- K4: selective scan + dt_proj/softplus + skip + gate -> yg (f16) ---------
__global__ void mamba_k4_scan(const _Float16* __restrict__ xt,
                              const _Float16* __restrict__ zb,
                              const _Float16* __restrict__ Bm,
                              const _Float16* __restrict__ Cm,
                              const float* __restrict__ dt2,
                              const float* __restrict__ dtw,   // (64,2)
                              const float* __restrict__ dtb,   // (64,)
                              const float* __restrict__ A_log, // (64,16)
                              const float* __restrict__ Dp,    // (64,)
                              _Float16* __restrict__ yg) {
    const int idx = blockIdx.x * blockDim.x + threadIdx.x;     // bb*64 + d
    const int bb = idx >> 6, d = idx & 63;
    float A[D_STATE], h[D_STATE];
    #pragma unroll
    for (int n = 0; n < D_STATE; ++n) {
        A[n] = -__expf(A_log[d * 16 + n]);
        h[n] = 0.f;
    }
    const float w0 = dtw[d * 2 + 0], w1 = dtw[d * 2 + 1];
    const float db = dtb[d], Dd = Dp[d];
    for (int t = 0; t < 64; ++t) {
        const long tok = (long)t * 2048 + bb;
        float dl = softplus_f(dt2[tok * 2] * w0 + dt2[tok * 2 + 1] * w1 + db);
        float u  = (float)xt[tok * 64 + d];
        float du = dl * u;
        v8h bv0 = *(const v8h*)&Bm[tok * 16];
        v8h bv1 = *(const v8h*)&Bm[tok * 16 + 8];
        v8h cv0 = *(const v8h*)&Cm[tok * 16];
        v8h cv1 = *(const v8h*)&Cm[tok * 16 + 8];
        float y = 0.f;
        #pragma unroll
        for (int n = 0; n < 8; ++n) {
            h[n] = __expf(dl * A[n]) * h[n] + du * (float)bv0[n];
            y += h[n] * (float)cv0[n];
        }
        #pragma unroll
        for (int n = 0; n < 8; ++n) {
            h[8 + n] = __expf(dl * A[8 + n]) * h[8 + n] + du * (float)bv1[n];
            y += h[8 + n] * (float)cv1[n];
        }
        float zz = (float)zb[tok * 64 + d];
        yg[tok * 64 + d] = (_Float16)((y + u * Dd) * silu_f(zz));
    }
}

// ---- K5: out_proj WMMA + head + coalesced scatter to (B,S,H,W) ---------------
__global__ void mamba_k5_head(const _Float16* __restrict__ yg,
                              const float* __restrict__ ow,   // (32,64)
                              const float* __restrict__ l1w,  // (16,32)
                              const float* __restrict__ l1b,  // (16,)
                              const float* __restrict__ l2w,  // (1,16)
                              const float* __restrict__ l2b,  // (1,)
                              float* __restrict__ out) {
    __shared__ float tile[8][16][33];
    __shared__ float mid[8][16][17];
    const int wv   = threadIdx.x >> 5;
    const int lane = threadIdx.x & 31;
    const int m16  = lane & 15;
    const int hi   = lane >> 4;
    const int kb   = hi ? 8 : 0;
    const int tidx = blockIdx.x * 8 + wv;          // tile 0..8191
    const long token0 = (long)tidx * 16;

    v16h a0, a1;
    #pragma unroll
    for (int e = 0; e < 16; ++e) {
        int k = frag_k(kb, e);
        a0[e] = yg[(token0 + m16) * 64 + k];
        a1[e] = yg[(token0 + m16) * 64 + 32 + k];
    }
    #pragma unroll
    for (int j = 0; j < 2; ++j) {                  // N = 32 -> 2 tiles
        v16h b0, b1;
        #pragma unroll
        for (int e = 0; e < 16; ++e) {
            int k  = frag_k(kb, e);
            int ng = j * 16 + m16;
            b0[e] = (_Float16)ow[ng * 64 + k];
            b1[e] = (_Float16)ow[ng * 64 + 32 + k];
        }
        v8f c = {};
        c = __builtin_amdgcn_wmma_f32_16x16x32_f16(false, a0, false, b0,
                                                   (short)0, c, false, false);
        c = __builtin_amdgcn_wmma_f32_16x16x32_f16(false, a1, false, b1,
                                                   (short)0, c, false, false);
        #pragma unroll
        for (int v = 0; v < 8; ++v) {
            int mm = v + hi * 8;
            tile[wv][mm][j * 16 + m16] = c[v];
        }
    }
    __syncthreads();
    // lin1: 16 tokens x 16 outputs, 8 per lane
    for (int idx = lane; idx < 256; idx += 32) {
        int mm = idx >> 4, n = idx & 15;
        float acc = l1b[n];
        #pragma unroll
        for (int k = 0; k < 32; ++k) acc += tile[wv][mm][k] * l1w[n * 32 + k];
        mid[wv][mm][n] = acc;
    }
    __syncthreads();
    // lin2 + scatter; 16 consecutive bb at fixed s -> contiguous out addresses
    if (lane < 16) {
        int mm = lane;
        float acc = l2b[0];
        #pragma unroll
        for (int n = 0; n < 16; ++n) acc += mid[wv][mm][n] * l2w[n];
        long token = token0 + mm;
        int s  = (int)(token >> 11);
        int bbi = (int)(token & 2047);
        int b  = bbi >> 10;
        int hw = bbi & 1023;
        out[(long)(b * 64 + s) * 1024 + hw] = acc;
    }
}

extern "C" void kernel_launch(void* const* d_in, const int* in_sizes, int n_in,
                              void* d_out, int out_size, void* d_ws, size_t ws_size,
                              hipStream_t stream) {
    (void)in_sizes; (void)n_in; (void)out_size; (void)ws_size;
    const float* x        = (const float*)d_in[0];
    const float* in_w     = (const float*)d_in[1];
    const float* conv_w   = (const float*)d_in[2];
    const float* conv_b   = (const float*)d_in[3];
    const float* xproj_w  = (const float*)d_in[4];
    const float* dtproj_w = (const float*)d_in[5];
    const float* dtproj_b = (const float*)d_in[6];
    const float* A_log    = (const float*)d_in[7];
    const float* Dp       = (const float*)d_in[8];
    const float* out_w    = (const float*)d_in[9];
    const float* l1w      = (const float*)d_in[10];
    const float* l1b      = (const float*)d_in[11];
    const float* l2w      = (const float*)d_in[12];
    const float* l2b      = (const float*)d_in[13];
    float* out = (float*)d_out;

    const long T = T_TOKENS;
    _Float16* xh  = (_Float16*)d_ws;        // [T,64]
    _Float16* zb  = xh + T * 64;            // [T,64]
    _Float16* xt  = zb + T * 64;            // [T,64]
    _Float16* Bm  = xt + T * 64;            // [T,16]
    _Float16* Cm  = Bm + T * 16;            // [T,16]
    float*    dt2 = (float*)(Cm + T * 16);  // [T,2]
    _Float16* yg  = xh;                     // reuse xh (dead after K2)

    mamba_k1_inproj<<<256,  256, 0, stream>>>(x, in_w, xh, zb);      // 2048 waves x 4 tiles
    mamba_k2_conv  <<<512,  256, 0, stream>>>(xh, conv_w, conv_b, xt);
    mamba_k3_xproj <<<1024, 256, 0, stream>>>(xt, xproj_w, Bm, Cm, dt2);
    mamba_k4_scan  <<<512,  256, 0, stream>>>(xt, zb, Bm, Cm, dt2,
                                              dtproj_w, dtproj_b, A_log, Dp, yg);
    mamba_k5_head  <<<1024, 256, 0, stream>>>(yg, out_w, l1w, l1b, l2w, l2b, out);
}
